// AttentionToken_37477884625643
// MI455X (gfx1250) — compile-verified, matrix operand order checked by reading, not run on hardware
//
#include <hip/hip_runtime.h>
#include <hip/hip_bf16.h>

typedef __attribute__((ext_vector_type(16))) _Float16 v16h;
typedef __attribute__((ext_vector_type(8)))  float    v8f;

#define SEQ   512
#define BATCH 2048
#define EMB   30
#define NTOK  (SEQ * BATCH)          // 1,048,576 tokens
#define NELEM ((long)NTOK * EMB)     // 31,457,280 f32
#define WELEM (EMB * EMB)            // 900

// Branch-free tanh: hardware v_tanh_f32 on gfx1250 if the builtin exists,
// else 1 - 2/(exp(2x)+1) (single v_exp_f32, saturates correctly at +/-inf).
__device__ __forceinline__ float fast_tanh(float v) {
#if __has_builtin(__builtin_amdgcn_tanhf)
    return __builtin_amdgcn_tanhf(v);
#else
    const float e = __expf(2.0f * v);
    return 1.0f - 2.0f / (e + 1.0f);
#endif
}

// ---------------------------------------------------------------------------
// Phase 1: per-token logits via WMMA.
// One wave32 = one 16-token M-tile. A:16x32 f16 (K=30 zero-padded), B: two
// 32x16 tiles of W (N=30 zero-padded). 2x v_wmma_f32_16x16x32_f16 per tile.
// All loads unconditional (clamped indices); padding zeroed by value-selects
// so EXEC stays all-1s (WMMA requirement) and no saveexec blocks are emitted.
// ---------------------------------------------------------------------------
__global__ __launch_bounds__(256) void attn_logits_kernel(
    const float* __restrict__ x,     // (S,B,E)
    const float* __restrict__ W,     // (E,E)
    const float* __restrict__ bias,  // (E,1)
    const float* __restrict__ proj,  // (E,1)
    float* __restrict__ logits)      // (B,S)
{
    const int lane   = threadIdx.x & 31;
    const int wave   = threadIdx.x >> 5;
    const int half16 = lane >> 4;    // 0: lanes 0-15, 1: lanes 16-31
    const int l      = lane & 15;

    const long tile = (long)blockIdx.x * 8 + wave;   // 16 tokens per tile
    const long t0   = tile * 16;

    const int n0 = l;        // B-tile-0 column this lane owns (always valid)
    const int n1 = 16 + l;   // B-tile-1 column (30/31 are padding)

    // ---- B tiles from W (K = e, N = f); ISA 16-bit K striping ----
    v16h b0, b1;
#pragma unroll
    for (int h = 0; h < 16; ++h) {
        const int k = (h < 8) ? ((half16 ? 8 : 0) + h)
                              : ((half16 ? 24 : 16) + (h - 8));
        int i0 = k * EMB + n0;  i0 = (i0 < WELEM) ? i0 : (WELEM - 1);
        int i1 = k * EMB + n1;  i1 = (i1 < WELEM) ? i1 : (WELEM - 1);
        float w0 = W[i0];                         // unconditional loads
        float w1 = W[i1];
        w0 = (k < EMB) ? w0 : 0.0f;               // value masks -> v_cndmask
        w1 = (k < EMB && n1 < EMB) ? w1 : 0.0f;
        b0[h] = (_Float16)w0;
        b1[h] = (_Float16)w1;
    }

    // ---- A tile: 16 tokens x 32 K (two contiguous 8-float runs per lane) ----
    const long rowbase = (t0 + l) * EMB;          // always even -> 8B aligned
    const int  kA = half16 ? 8  : 0;
    const int  kB = half16 ? 24 : 16;
    float va[8], vb[8];
#pragma unroll
    for (int j = 0; j < 4; ++j) {
        const long iA = rowbase + kA + 2 * j;
        long       iB = rowbase + kB + 2 * j;
        iB = (iB <= NELEM - 2) ? iB : (NELEM - 2);      // clamp (no branch)
        const float2 fa = *(const float2*)(x + iA);
        const float2 fb = *(const float2*)(x + iB);
        va[2 * j]     = fa.x;  va[2 * j + 1] = fa.y;
        vb[2 * j]     = fb.x;  vb[2 * j + 1] = fb.y;
    }
    // only k=30,31 (half-wave 1, last float2 of run B) are padding
    vb[6] = half16 ? 0.0f : vb[6];
    vb[7] = half16 ? 0.0f : vb[7];

    v16h a;
#pragma unroll
    for (int j = 0; j < 8; ++j) {
        a[j]     = (_Float16)va[j];
        a[8 + j] = (_Float16)vb[j];
    }

    v8f c0 = {}, c1 = {};
    c0 = __builtin_amdgcn_wmma_f32_16x16x32_f16(false, a, false, b0,
                                                (short)0, c0, false, false);
    c1 = __builtin_amdgcn_wmma_f32_16x16x32_f16(false, a, false, b1,
                                                (short)0, c1, false, false);

    // ---- bias + tanh, scale by proj[n]; per-lane partial of the proj dot ----
    const float bia0 = bias[n0];
    const float bia1 = bias[(n1 < EMB) ? n1 : (EMB - 1)];
    const float p0   = proj[n0];
    float       p1   = proj[(n1 < EMB) ? n1 : (EMB - 1)];
    p1 = (n1 < EMB) ? p1 : 0.0f;                  // kill padded columns

    float part[8];
#pragma unroll
    for (int r = 0; r < 8; ++r) {
        const float s0 = fast_tanh(c0[r] + bia0);
        const float s1 = fast_tanh(c1[r] + bia1);
        part[r] = s0 * p0 + s1 * p1;
    }

    // ---- sum over N: reduce across the 16 lanes of each half-wave ----
#pragma unroll
    for (int off = 8; off > 0; off >>= 1) {
#pragma unroll
        for (int r = 0; r < 8; ++r)
            part[r] += __shfl_xor(part[r], off, 16);
    }

    // C/D layout: VGPR r holds row r (lanes 0-15) / row 8+r (lanes 16-31)
    if (l == 0) {
#pragma unroll
        for (int r = 0; r < 8; ++r) {
            const long tok = t0 + half16 * 8 + r;   // tok = s*BATCH + b
            const long s   = tok >> 11;             // / 2048
            const long b   = tok & (BATCH - 1);
            logits[b * SEQ + s] = fast_tanh(part[r]);
        }
    }
}

// ---------------------------------------------------------------------------
// Phase 2: softmax over SEQ=512 per batch row, in place. One block per batch.
// ---------------------------------------------------------------------------
__global__ __launch_bounds__(256) void softmax_kernel(float* __restrict__ logits)
{
    __shared__ float red[256];
    float* row = logits + (long)blockIdx.x * SEQ;
    const int tid = threadIdx.x;

    const float v0 = row[tid];
    const float v1 = row[tid + 256];

    red[tid] = fmaxf(v0, v1);
    __syncthreads();
#pragma unroll
    for (int s = 128; s > 0; s >>= 1) {
        if (tid < s) red[tid] = fmaxf(red[tid], red[tid + s]);
        __syncthreads();
    }
    const float mx = red[0];
    __syncthreads();

    const float e0 = __expf(v0 - mx);
    const float e1 = __expf(v1 - mx);
    red[tid] = e0 + e1;
    __syncthreads();
#pragma unroll
    for (int s = 128; s > 0; s >>= 1) {
        if (tid < s) red[tid] += red[tid + s];
        __syncthreads();
    }
    const float inv = 1.0f / red[0];

    row[tid]       = e0 * inv;
    row[tid + 256] = e1 * inv;
}

// ---------------------------------------------------------------------------
// Phase 3: emit (B,2,S,E): plane 0 = transposed input, plane 1 = scaled.
// Indexed by output layout so both 30-float token segments store contiguously.
// ---------------------------------------------------------------------------
__global__ __launch_bounds__(256) void scatter_kernel(
    const float* __restrict__ x,      // (S,B,E)
    const float* __restrict__ wgt,    // (B,S) softmax weights
    float* __restrict__ out)          // (B,2,S,E)
{
    const long i = (long)blockIdx.x * blockDim.x + threadIdx.x;  // over B*S*E
    if (i >= NELEM) return;

    const int  e  = (int)(i % EMB);
    const long bs = i / EMB;                 // b*SEQ + s
    const int  s  = (int)(bs & (SEQ - 1));
    const int  b  = (int)(bs >> 9);          // / 512

    const float* src = x + ((long)s * BATCH + b) * EMB;
    __builtin_prefetch(src + 8 * BATCH * EMB, 0, 0);   // global_prefetch hint
    const float xv = src[e];
    const float wv = wgt[bs];

    out[((long)(b * 2)     * SEQ + s) * EMB + e] = xv;
    out[((long)(b * 2 + 1) * SEQ + s) * EMB + e] = xv * wv;
}

extern "C" void kernel_launch(void* const* d_in, const int* in_sizes, int n_in,
                              void* d_out, int out_size, void* d_ws, size_t ws_size,
                              hipStream_t stream)
{
    const float* x    = (const float*)d_in[0];   // (512,2048,30)
    const float* W    = (const float*)d_in[1];   // (30,30)
    const float* bias = (const float*)d_in[2];   // (30,1)
    const float* proj = (const float*)d_in[3];   // (30,1)
    float* out        = (float*)d_out;           // (2048,2,512,30)
    float* logits     = (float*)d_ws;            // B*S floats = 4 MB

    // 1,048,576 tokens / 16 per wave / 8 waves per block = 8192 blocks
    attn_logits_kernel<<<NTOK / (16 * 8), 256, 0, stream>>>(x, W, bias, proj, logits);
    softmax_kernel<<<BATCH, 256, 0, stream>>>(logits);
    scatter_kernel<<<(int)((NELEM + 255) / 256), 256, 0, stream>>>(x, logits, out);
}
